// MemoryReader_73890617360687
// MI455X (gfx1250) — matrix operand order repaired and checked by reading, not compiled, not run on hardware
//
#include <hip/hip_runtime.h>

// ---------------- problem constants ----------------
constexpr int kB   = 32;
constexpr int kM   = 4096;
constexpr int kE   = 1024;
constexpr int kQ   = 64;
constexpr int kH   = 16;
constexpr int kCTX = 128;
constexpr int kDH  = 64;          // kE / kH
#define QSCALE 0.125f             // 1/sqrt(DH)

// ---------------- WMMA helper types ----------------
typedef __attribute__((ext_vector_type(16))) __bf16 bf16x16;
typedef __attribute__((ext_vector_type(8)))  __bf16 bf16x8;
typedef __attribute__((ext_vector_type(8)))  float  f32x8;

// ---------------- TDM availability ----------------
#if defined(__HIP_DEVICE_COMPILE__) && \
    __has_builtin(__builtin_amdgcn_tensor_load_to_lds) && \
    __has_builtin(__builtin_amdgcn_s_wait_tensorcnt)
#define USE_TDM 1
typedef __attribute__((ext_vector_type(4))) unsigned int u32x4;
typedef __attribute__((ext_vector_type(4))) int          i32x4;
typedef __attribute__((ext_vector_type(8))) int          i32x8;
#else
#define USE_TDM 0
#endif

__device__ __forceinline__ unsigned short f2bf(float f) {
  unsigned int u = __float_as_uint(f);
  unsigned int r = (u + 0x7FFFu + ((u >> 16) & 1u)) >> 16;   // RNE
  return (unsigned short)r;
}
__device__ __forceinline__ float bf2f(unsigned short s) {
  return __uint_as_float(((unsigned int)s) << 16);
}

// Build a 16-element bf16 fragment from two contiguous 8-element chunks.
// Matches ISA A-layout: lane holds K = {half*8..half*8+7, 16+half*8..16+half*8+7}
__device__ __forceinline__ bf16x16 load_frag(const unsigned short* p0,
                                             const unsigned short* p1) {
  bf16x8 lo = *reinterpret_cast<const bf16x8*>(p0);
  bf16x8 hi = *reinterpret_cast<const bf16x8*>(p1);
  return __builtin_shufflevector(lo, hi, 0,1,2,3,4,5,6,7,8,9,10,11,12,13,14,15);
}

__device__ __forceinline__ f32x8 wmma_bf16(bf16x16 a, bf16x16 b, f32x8 c) {
  return __builtin_amdgcn_wmma_f32_16x16x32_bf16(false, a, false, b,
                                                 (short)0, c, false, false);
}

#if USE_TDM
// Issue a TDM load of a 32(m) x 64(dh) bf16 tile from V (row-major, ld=64)
// into LDS at byte offset lds_off. D# per CDNA5 ISA.
__device__ __forceinline__ void tdm_load_v_tile(const unsigned short* gsrc,
                                                unsigned lds_off) {
  const unsigned long long ga = (unsigned long long)gsrc;
  u32x4 g0;
  g0[0] = 1u;                                                  // count=1 (valid user D#)
  g0[1] = lds_off;                                             // lds_addr (bytes)
  g0[2] = (unsigned)ga;                                        // global_addr[31:0]
  g0[3] = (unsigned)((ga >> 32) & 0x01FFFFFFu) | (2u << 30);   // addr[56:32] | type=2
  i32x8 g1;
  g1[0] = 0x00010000;          // data_size=2 bytes; no multicast/pad/iterate
  g1[1] = (int)(64u << 16);    // tensor_dim0 = 64 (dh)          -> bits[79:48] lo
  g1[2] = (int)(4096u << 16);  // tensor_dim1 = 4096 (m rows)    -> bits[111:80] lo
  g1[3] = (int)(64u << 16);    // tile_dim0 = 64                 -> bits[127:112]
  g1[4] = 32;                  // tile_dim1 = 32, tile_dim2 = 0
  g1[5] = 64;                  // tensor_dim0_stride = 64 elements
  g1[6] = 0;
  g1[7] = 0;
  const i32x4 z4 = {0, 0, 0, 0};
#if __clang_major__ >= 23
  const i32x8 z8 = {0, 0, 0, 0, 0, 0, 0, 0};
  __builtin_amdgcn_tensor_load_to_lds(g0, g1, z4, z4, z8, 0);
#else
  __builtin_amdgcn_tensor_load_to_lds(g0, g1, z4, z4, 0);
#endif
}
#endif

// =====================================================================
// Kernel 0: pooled = mean(context), proj = pooled@ctx_w.T + ctx_b,
//           queries = queries_p + proj  (fp32 out + bf16 copy),
//           gates = sigmoid(pooled@gate_w.T + gate_b)
// =====================================================================
__global__ __launch_bounds__(256)
void prep_kernel(const float* __restrict__ context, const float* __restrict__ queries_p,
                 const float* __restrict__ ctx_w, const float* __restrict__ ctx_b,
                 const float* __restrict__ gate_w, const float* __restrict__ gate_b,
                 float* __restrict__ out_queries, float* __restrict__ out_gates,
                 unsigned short* __restrict__ q_bf16) {
  __shared__ float pooled[kE];
  __shared__ float proj[kE];
  const int b = blockIdx.x, t = threadIdx.x;

  for (int e = t; e < kE; e += 256) {
    float s = 0.f;
    const float* cp = context + ((long)b * kCTX) * kE + e;
    for (int c = 0; c < kCTX; ++c) s += cp[(long)c * kE];
    pooled[e] = s * (1.f / kCTX);
  }
  __syncthreads();

  for (int e = t; e < kE; e += 256) {
    float s = ctx_b[e];
    const float* wr = ctx_w + (long)e * kE;
    for (int j = 0; j < kE; ++j) s += pooled[j] * wr[j];
    proj[e] = s;
  }
  if (t < kQ) {
    float s = gate_b[t];
    const float* wr = gate_w + (long)t * kE;
    for (int j = 0; j < kE; ++j) s += pooled[j] * wr[j];
    out_gates[b * kQ + t] = 1.f / (1.f + __expf(-s));
  }
  __syncthreads();

  for (int idx = t; idx < kQ * kE; idx += 256) {
    const int q = idx >> 10, e = idx & (kE - 1);
    const float v = queries_p[idx] + proj[e];
    const long o = ((long)b * kQ + q) * kE + e;
    out_queries[o] = v;
    q_bf16[o] = f2bf(v);
  }
}

// =====================================================================
// Generic bf16 WMMA GEMM:  C[m,n] = alpha * (sum_k A[m,k]*W[n,k] + bias[n])
// A: Mrows x 1024 (bf16 if AMODE==0, fp32 if AMODE==1), W: fp32 (N x 1024)
// Block tile 64x64, 4 waves, each wave = 16 rows x 64 cols. K step 32.
// OMODE: 0 -> qh (B,H,Q,DH) bf16   1 -> K (B,H,DH,M) bf16
//        2 -> V  (B,H,M,DH) bf16   3 -> fp32 row-major (ldc=1024)
// =====================================================================
template <int AMODE, int OMODE>
__global__ __launch_bounds__(128)
void gemm_kernel(const void* __restrict__ Av, const float* __restrict__ W,
                 const float* __restrict__ bias, void* __restrict__ Out,
                 float alpha) {
  constexpr int K = kE;
  constexpr int LAS = 40;  // lA row stride (ushorts), 80B = 16B-aligned
  constexpr int LBS = 72;  // lB row stride (ushorts), 144B = 16B-aligned
  __shared__ unsigned short lA[64 * LAS];
  __shared__ unsigned short lB[32 * LBS];

  const int m0 = blockIdx.x * 64;
  const int n0 = blockIdx.y * 64;
  const int t = threadIdx.x;
  const int wv = t >> 5, lane = t & 31, half = lane >> 4, rl = lane & 15;

  f32x8 acc[4] = {};

  for (int k0 = 0; k0 < K; k0 += 32) {
    {  // stage A tile: 64 rows x 32 k (bf16 in LDS)
      const int r = t >> 1, c = (t & 1) * 16;
      if (AMODE == 0) {
        const unsigned short* src =
            (const unsigned short*)Av + (long)(m0 + r) * K + k0 + c;
        uint4 v0 = ((const uint4*)src)[0];
        uint4 v1 = ((const uint4*)src)[1];
        *(uint4*)&lA[r * LAS + c] = v0;
        *(uint4*)&lA[r * LAS + c + 8] = v1;
      } else {
        const float* src = (const float*)Av + (long)(m0 + r) * K + k0 + c;
        __builtin_prefetch(src + 32, 0, 1);   // next k-slice -> global_prefetch_b8
#pragma unroll
        for (int i = 0; i < 16; ++i) lA[r * LAS + c + i] = f2bf(src[i]);
      }
    }
    {  // stage W tile transposed into B layout: 32 k-rows x 64 n-cols
      const int n = t >> 1, c = (t & 1) * 16;
      const float* src = W + (long)(n0 + n) * K + k0 + c;
#pragma unroll
      for (int i = 0; i < 16; ++i) lB[(c + i) * LBS + n] = f2bf(src[i]);
    }
    __syncthreads();

    const unsigned short* abase = &lA[(wv * 16 + rl) * LAS];
    bf16x16 a = load_frag(abase + half * 8, abase + 16 + half * 8);
#pragma unroll
    for (int j = 0; j < 4; ++j) {
      bf16x16 bb;
#pragma unroll
      for (int s = 0; s < 16; ++s) {
        const int kk = (s < 8) ? half * 8 + s : 16 + half * 8 + (s - 8);
        bb[s] = __builtin_bit_cast(__bf16, lB[kk * LBS + j * 16 + rl]);
      }
      acc[j] = wmma_bf16(a, bb, acc[j]);
    }
    __syncthreads();
  }

  // epilogue: C layout -> lane holds col rl, rows i + 8*half
#pragma unroll
  for (int j = 0; j < 4; ++j) {
    const int ncol = n0 + j * 16 + rl;
    const float bv = bias ? bias[ncol] : 0.f;
#pragma unroll
    for (int i = 0; i < 8; ++i) {
      const int mrow = m0 + wv * 16 + i + 8 * half;
      const float v = (acc[j][i] + bv) * alpha;
      if (OMODE == 0) {        // qh: (B,H,Q,DH)
        const int b = mrow >> 6, q = mrow & 63, hh = ncol >> 6, d = ncol & 63;
        ((unsigned short*)Out)[(((long)(b * kH + hh) * kQ + q) * kDH) + d] = f2bf(v);
      } else if (OMODE == 1) { // K: (B,H,DH,M)
        const int b = mrow >> 12, mm = mrow & 4095, hh = ncol >> 6, d = ncol & 63;
        ((unsigned short*)Out)[((long)(b * kH + hh) * kDH + d) * kM + mm] = f2bf(v);
      } else if (OMODE == 2) { // V: (B,H,M,DH)
        const int b = mrow >> 12, mm = mrow & 4095, hh = ncol >> 6, d = ncol & 63;
        ((unsigned short*)Out)[((long)(b * kH + hh) * kM + mm) * kDH + d] = f2bf(v);
      } else {                 // fp32 row-major
        ((float*)Out)[(long)mrow * kE + ncol] = v;
      }
    }
  }
}

// =====================================================================
// Attention scores + two-pass online softmax. One block per (b,h).
// =====================================================================
__device__ __forceinline__ void score_tile(const unsigned short* Kbase, int m0,
                                           int lane, bf16x16 aq0, bf16x16 aq1,
                                           f32x8 acc[4]) {
#pragma unroll
  for (int j = 0; j < 4; ++j) {
    bf16x16 b0, b1;
#pragma unroll
    for (int s = 0; s < 16; ++s) {
      const int kk = (s < 8) ? (lane >> 4) * 8 + s : 16 + (lane >> 4) * 8 + (s - 8);
      b0[s] = __builtin_bit_cast(__bf16, Kbase[(long)kk * kM + m0 + j * 16 + (lane & 15)]);
      b1[s] = __builtin_bit_cast(__bf16, Kbase[(long)(32 + kk) * kM + m0 + j * 16 + (lane & 15)]);
    }
    acc[j] = wmma_bf16(aq0, b0, acc[j]);
    acc[j] = wmma_bf16(aq1, b1, acc[j]);
  }
}

__global__ __launch_bounds__(128)
void attn_score_kernel(const unsigned short* __restrict__ qh,
                       const unsigned short* __restrict__ Kw,
                       unsigned short* __restrict__ Pw) {
  const int bh = blockIdx.x;
  const int t = threadIdx.x, wv = t >> 5, lane = t & 31, half = lane >> 4, rl = lane & 15;
  const unsigned short* qbase = qh + (long)bh * (kQ * kDH);
  const unsigned short* Kbase = Kw + (long)bh * (kDH * kM);
  unsigned short* Pbase = Pw + (long)bh * ((long)kQ * kM);

  const unsigned short* qrow = qbase + (wv * 16 + rl) * kDH;
  const bf16x16 aq0 = load_frag(qrow + half * 8, qrow + 16 + half * 8);
  const bf16x16 aq1 = load_frag(qrow + 32 + half * 8, qrow + 48 + half * 8);

  float runmax[8], runsum[8];
#pragma unroll
  for (int i = 0; i < 8; ++i) { runmax[i] = -1e30f; runsum[i] = 0.f; }

  // ---- pass 1: stats ----
  for (int m0 = 0; m0 < kM; m0 += 64) {
    f32x8 acc[4] = {};
    score_tile(Kbase, m0, lane, aq0, aq1, acc);
#pragma unroll
    for (int i = 0; i < 8; ++i) {
      float tm = fmaxf(fmaxf(acc[0][i], acc[1][i]), fmaxf(acc[2][i], acc[3][i]));
#pragma unroll
      for (int s = 1; s < 16; s <<= 1) tm = fmaxf(tm, __shfl_xor(tm, s, 32));
      const float nm = fmaxf(runmax[i], tm);
      float ts = 0.f;
#pragma unroll
      for (int j = 0; j < 4; ++j) ts += __expf(acc[j][i] - nm);
#pragma unroll
      for (int s = 1; s < 16; s <<= 1) ts += __shfl_xor(ts, s, 32);
      runsum[i] = runsum[i] * __expf(runmax[i] - nm) + ts;
      runmax[i] = nm;
    }
  }
  float inv[8];
#pragma unroll
  for (int i = 0; i < 8; ++i) inv[i] = 1.f / runsum[i];

  // ---- pass 2: recompute, write normalized probs ----
  for (int m0 = 0; m0 < kM; m0 += 64) {
    f32x8 acc[4] = {};
    score_tile(Kbase, m0, lane, aq0, aq1, acc);
#pragma unroll
    for (int j = 0; j < 4; ++j) {
      const int col = m0 + j * 16 + rl;
#pragma unroll
      for (int i = 0; i < 8; ++i) {
        const int qr = wv * 16 + i + 8 * half;
        const float p = __expf(acc[j][i] - runmax[i]) * inv[i];
        Pbase[(long)qr * kM + col] = f2bf(p);
      }
    }
  }
}

// =====================================================================
// attention output = mean over heads of probs
// =====================================================================
__global__ __launch_bounds__(256)
void attn_mean_kernel(const unsigned short* __restrict__ Pw,
                      float* __restrict__ out_attn) {
  const int idx = blockIdx.x * 256 + threadIdx.x;  // < B*Q*M = 8388608
  const int b = idx / (kQ * kM);
  const int r = idx % (kQ * kM);
  float s = 0.f;
#pragma unroll
  for (int h = 0; h < kH; ++h)
    s += bf2f(Pw[(long)(b * kH + h) * (kQ * kM) + r]);
  out_attn[idx] = s * (1.f / kH);
}

// =====================================================================
// attn_out = probs @ V, one block per (b,h), 4 waves x 16 q-rows.
// V tiles (32 x 64 bf16, 4KB) are DMA'd into LDS by the Tensor Data Mover,
// double-buffered and tracked with TENSORcnt; falls back to direct global
// loads when the TDM builtin is unavailable.
// =====================================================================
__global__ __launch_bounds__(128)
void attn_pv_kernel(const unsigned short* __restrict__ Pw,
                    const unsigned short* __restrict__ Vw,
                    unsigned short* __restrict__ AO) {
  const int bh = blockIdx.x;
  const int b = bh >> 4, h = bh & 15;
  const int t = threadIdx.x, wv = t >> 5, lane = t & 31, half = lane >> 4, rl = lane & 15;
  const unsigned short* Prow =
      Pw + (long)bh * ((long)kQ * kM) + (long)(wv * 16 + rl) * kM;
  const unsigned short* Vbase = Vw + (long)bh * (kM * kDH);

  f32x8 acc[4] = {};

#if USE_TDM
  __shared__ unsigned short vtile[2][32 * kDH];   // double-buffered 4KB tiles
  __shared__ unsigned short vscr[32 * kDH];       // tail dummy target
  const unsigned ldsbuf[2] = {
      (unsigned)(unsigned long long)(const void*)&vtile[0][0],
      (unsigned)(unsigned long long)(const void*)&vtile[1][0]};
  const unsigned ldsscr = (unsigned)(unsigned long long)(const void*)&vscr[0];

  if (wv == 0) {
    tdm_load_v_tile(Vbase, ldsbuf[0]);            // tile 0
    tdm_load_v_tile(Vbase + 32 * kDH, ldsbuf[1]); // tile 1
  }
  for (int i = 0; i < kM / 32; ++i) {
    const int m0 = i * 32;
    if (wv == 0) __builtin_amdgcn_s_wait_tensorcnt(1);  // tile i complete
    __syncthreads();
    asm volatile("" ::: "memory");
    const unsigned short* vt = &vtile[i & 1][0];
    bf16x16 a = load_frag(Prow + m0 + half * 8, Prow + m0 + 16 + half * 8);
#pragma unroll
    for (int j = 0; j < 4; ++j) {
      bf16x16 bb;
#pragma unroll
      for (int s = 0; s < 16; ++s) {
        const int kk = (s < 8) ? half * 8 + s : 16 + half * 8 + (s - 8);
        bb[s] = __builtin_bit_cast(__bf16, vt[kk * kDH + j * 16 + rl]);
      }
      acc[j] = wmma_bf16(a, bb, acc[j]);
    }
    __syncthreads();   // everyone done with buf[i&1] before it is refilled
    if (wv == 0) {
      if (i + 2 < kM / 32)
        tdm_load_v_tile(Vbase + (long)(m0 + 64) * kDH, ldsbuf[i & 1]);
      else if (i + 2 == kM / 32)
        tdm_load_v_tile(Vbase, ldsscr);  // keep pipeline depth so wait(1) frees tile 127
    }
  }
#else
  for (int m0 = 0; m0 < kM; m0 += 32) {
    bf16x16 a = load_frag(Prow + m0 + half * 8, Prow + m0 + 16 + half * 8);
#pragma unroll
    for (int j = 0; j < 4; ++j) {
      bf16x16 bb;
#pragma unroll
      for (int s = 0; s < 16; ++s) {
        const int kk = (s < 8) ? half * 8 + s : 16 + half * 8 + (s - 8);
        bb[s] = __builtin_bit_cast(__bf16, Vbase[(long)(m0 + kk) * kDH + j * 16 + rl]);
      }
      acc[j] = wmma_bf16(a, bb, acc[j]);
    }
  }
#endif

#pragma unroll
  for (int j = 0; j < 4; ++j)
#pragma unroll
    for (int i = 0; i < 8; ++i) {
      const int q = wv * 16 + i + 8 * half;
      const int d = j * 16 + rl;
      AO[((long)(b * kQ + q)) * kE + h * kDH + d] = f2bf(acc[j][i]);
    }
}

// =====================================================================
// gated LayerNorm + residual epilogue, one block per (b,q) row
// =====================================================================
__global__ __launch_bounds__(256)
void final_kernel(const float* __restrict__ RO, const float* __restrict__ queries,
                  const float* __restrict__ gates, const float* __restrict__ ln_g,
                  const float* __restrict__ ln_b, float* __restrict__ out_normed) {
  __shared__ float red[256];
  __shared__ float red2[256];
  const int row = blockIdx.x, t = threadIdx.x;
  const int b = row >> 6, q = row & 63;
  const float g = gates[b * kQ + q];

  float vals[4];
  float s = 0.f, s2 = 0.f;
#pragma unroll
  for (int i = 0; i < 4; ++i) {
    const float v = RO[(long)row * kE + t + i * 256] * g;
    vals[i] = v; s += v; s2 += v * v;
  }
  red[t] = s; red2[t] = s2;
  __syncthreads();
  for (int off = 128; off > 0; off >>= 1) {
    if (t < off) { red[t] += red[t + off]; red2[t] += red2[t + off]; }
    __syncthreads();
  }
  const float mu = red[0] * (1.f / kE);
  const float var = red2[0] * (1.f / kE) - mu * mu;
  const float rstd = rsqrtf(var + 1e-5f);
#pragma unroll
  for (int i = 0; i < 4; ++i) {
    const int e = t + i * 256;
    const float nv = ln_g[e] * (vals[i] - mu) * rstd + ln_b[e] +
                     0.1f * queries[(long)row * kE + e] * g;
    out_normed[(long)row * kE + e] = nv;
  }
}

// =====================================================================
extern "C" void kernel_launch(void* const* d_in, const int* in_sizes, int n_in,
                              void* d_out, int out_size, void* d_ws, size_t ws_size,
                              hipStream_t stream) {
  (void)in_sizes; (void)n_in; (void)out_size; (void)ws_size;

  const float* memory    = (const float*)d_in[0];
  const float* context   = (const float*)d_in[1];
  const float* queries_p = (const float*)d_in[2];
  const float* ctx_w     = (const float*)d_in[3];
  const float* ctx_b     = (const float*)d_in[4];
  const float* in_proj_w = (const float*)d_in[5];
  const float* in_proj_b = (const float*)d_in[6];
  const float* out_w     = (const float*)d_in[7];
  const float* out_b     = (const float*)d_in[8];
  const float* ln_g      = (const float*)d_in[9];
  const float* ln_b      = (const float*)d_in[10];
  const float* gate_w    = (const float*)d_in[11];
  const float* gate_b    = (const float*)d_in[12];

  float* out_normed  = (float*)d_out;                       // (B,Q,E)
  float* out_attn    = out_normed + (size_t)kB * kQ * kE;   // (B,Q,M)
  float* out_gates   = out_attn + (size_t)kB * kQ * kM;     // (B,Q)
  float* out_queries = out_gates + (size_t)kB * kQ;         // (B,Q,E)

  char* ws = (char*)d_ws;
  size_t off = 0;
  auto alloc = [&](size_t bytes) {
    void* p = ws + off;
    off += (bytes + 255) & ~(size_t)255;
    return p;
  };
  unsigned short* q_bf16 = (unsigned short*)alloc((size_t)kB * kQ * kE * 2);
  unsigned short* qh_ws  = (unsigned short*)alloc((size_t)kB * kQ * kE * 2);
  unsigned short* K_ws   = (unsigned short*)alloc((size_t)kB * kM * kE * 2);
  unsigned short* V_ws   = (unsigned short*)alloc((size_t)kB * kM * kE * 2);
  unsigned short* P_ws   = (unsigned short*)alloc((size_t)kB * kH * kQ * kM * 2);
  unsigned short* AO_ws  = (unsigned short*)alloc((size_t)kB * kQ * kE * 2);
  float*          RO_ws  = (float*)alloc((size_t)kB * kQ * kE * 4);

  prep_kernel<<<kB, 256, 0, stream>>>(context, queries_p, ctx_w, ctx_b, gate_w,
                                      gate_b, out_queries, out_gates, q_bf16);

  dim3 gq(kB * kQ / 64, kE / 64);
  gemm_kernel<0, 0><<<gq, 128, 0, stream>>>(q_bf16, in_proj_w, in_proj_b, qh_ws, QSCALE);

  dim3 gkv(kB * kM / 64, kE / 64);
  gemm_kernel<1, 1><<<gkv, 128, 0, stream>>>(memory, in_proj_w + (size_t)kE * kE,
                                             in_proj_b + kE, K_ws, 1.f);
  gemm_kernel<1, 2><<<gkv, 128, 0, stream>>>(memory, in_proj_w + (size_t)2 * kE * kE,
                                             in_proj_b + 2 * kE, V_ws, 1.f);

  attn_score_kernel<<<kB * kH, 128, 0, stream>>>(qh_ws, K_ws, P_ws);
  attn_mean_kernel<<<(kB * kQ * kM) / 256, 256, 0, stream>>>(P_ws, out_attn);
  attn_pv_kernel<<<kB * kH, 128, 0, stream>>>(P_ws, V_ws, AO_ws);

  gemm_kernel<0, 3><<<gq, 128, 0, stream>>>(AO_ws, out_w, out_b, RO_ws, 1.f);

  final_kernel<<<kB * kQ, 256, 0, stream>>>(RO_ws, out_queries, out_gates, ln_g,
                                            ln_b, out_normed);
}
